// Sub4BitLinear_8521215115981
// MI455X (gfx1250) — matrix operand order repaired and checked by reading, not compiled, not run on hardware
//
#include <hip/hip_runtime.h>

typedef __attribute__((ext_vector_type(16))) __bf16   v16bf;
typedef __attribute__((ext_vector_type(2)))  __bf16   v2bf;
typedef __attribute__((ext_vector_type(8)))  float    v8f;
typedef __attribute__((ext_vector_type(4)))  float    vf4;
typedef __attribute__((ext_vector_type(2)))  unsigned v2u;
typedef __attribute__((ext_vector_type(4)))  unsigned v4u;

#define MDIM 8192   // BATCH*SEQ
#define NDIM 4096   // ROWS
#define KDIM 4096   // COLS
#define BM   256
#define BN   128
#define BK   32
#define LDX  40     // padded LDS row stride in halves (80B, keeps b128 16B-aligned)
#define LDW  40
#define NTHREADS 256

union FragU { v4u u[2]; v16bf f; };

__device__ __forceinline__ unsigned pack2(float lo, float hi) {
    v2bf h;
    h.x = (__bf16)lo;       // RTNE; pattern-matches v_cvt_pk_bf16_f32
    h.y = (__bf16)hi;
    return __builtin_bit_cast(unsigned, h);
}

__global__ __launch_bounds__(NTHREADS)
__attribute__((amdgpu_waves_per_eu(1)))
void sub4bit_linear_wmma(const float* __restrict__ x,
                         const float* __restrict__ quant_grid,
                         const int*   __restrict__ codes,
                         float* __restrict__ out) {
    __shared__ __align__(16) unsigned short xs[2][BM][LDX];  // 40 KB, x tiles (bf16)
    __shared__ __align__(16) unsigned short ws[2][BN][LDW];  // 20 KB, decoded W tiles
    __shared__ __align__(16) unsigned short gs[BN][16];      //  4 KB, per-row codebooks

    const int t    = threadIdx.x;
    const int lane = t & 31;
    const int wave = t >> 5;
    const int half = lane >> 4;
    const int l15  = lane & 15;

    const int block_n = blockIdx.x * BN;
    const int block_m = blockIdx.y * BM;

    // ---- preload codebook: 128 rows x 16 entries, fp32 -> bf16 ----
    {
        const int gr = t >> 1;
        const int ge = (t & 1) * 8;
        const float* src = quant_grid + (size_t)(block_n + gr) * 16 + ge;
        vf4 a = *(const vf4*)(src);
        vf4 b = *(const vf4*)(src + 4);
        v4u pk;
        pk.x = pack2(a.x, a.y);
        pk.y = pack2(a.z, a.w);
        pk.z = pack2(b.x, b.y);
        pk.w = pack2(b.z, b.w);
        *(v4u*)&gs[gr][ge] = pk;
    }
    __syncthreads();

    // wave tiling: 4 waves along M, 2 along N; each wave owns 64x64
    const int wm = (wave >> 1) * 64;
    const int wn = (wave & 1) * 64;

    // staging roles
    const int xr = t >> 3;          // x-stage row 0..31 (8 passes of 32 rows)
    const int xc = (t & 7) * 4;     // x-stage col (floats) 0..28
    const int wr = t >> 4;          // w-stage row 0..15 (8 passes of 16 rows)
    const int wc = t & 15;          // w-stage int32 index 0..15 (covers 32 K)

    const float* xp = x     + (size_t)(block_m + xr) * KDIM + xc;
    const int*   cp = codes + (size_t)(block_n + wr) * (KDIM / 2) + wc;

    // ---- prologue: stage K-step 0 into buffer 0 ----
#pragma unroll
    for (int p = 0; p < 8; ++p) {
        const vf4 v = *(const vf4*)(xp + (size_t)p * 32 * KDIM);
        v2u pk;
        pk.x = pack2(v.x, v.y);
        pk.y = pack2(v.z, v.w);
        *(v2u*)&xs[0][xr + p * 32][xc] = pk;
    }
#pragma unroll
    for (int p = 0; p < 8; ++p) {
        const int n = wr + p * 16;
        const int code = cp[(size_t)p * 16 * (KDIM / 2)] & 0xFF;
        *(unsigned*)&ws[0][n][wc * 2] =
            (unsigned)gs[n][code >> 4] | ((unsigned)gs[n][code & 15] << 16);
    }

    const float* xnext = xp + BK;
    const int*   cnext = cp + BK / 2;

    v8f acc[4][4];
#pragma unroll
    for (int i = 0; i < 4; ++i)
#pragma unroll
        for (int j = 0; j < 4; ++j)
            acc[i][j] = (v8f){0.f,0.f,0.f,0.f,0.f,0.f,0.f,0.f};

    __syncthreads();

    int buf = 0;
    for (int k0 = 0; k0 < KDIM; k0 += BK) {
        const bool has_next = (k0 + BK) < KDIM;

        // ---- issue next tile's global loads early (overlap with WMMA) ----
        vf4 xv[8];
        int cd[8];
        if (has_next) {
#pragma unroll
            for (int p = 0; p < 8; ++p)
                xv[p] = *(const vf4*)(xnext + (size_t)p * 32 * KDIM);
#pragma unroll
            for (int p = 0; p < 8; ++p)
                cd[p] = cnext[(size_t)p * 16 * (KDIM / 2)];
            if (k0 + 2 * BK < KDIM)
                __builtin_prefetch(xnext + BK, 0, 3);   // global_prefetch_b8
        }

        // ---- fragments per the 16-bit A/B VGPR layouts ----
        FragU a[4], b[4];
        const int ka = half * 8;    // A: lanes>=16 take K=8..15 / 24..31
        const int kb = half * 8;    // B: lanes>=16 take K=8..15 / 24..31
#pragma unroll
        for (int sm = 0; sm < 4; ++sm) {
            const int row = wm + sm * 16 + l15;
            a[sm].u[0] = *(const v4u*)&xs[buf][row][ka];
            a[sm].u[1] = *(const v4u*)&xs[buf][row][ka + 16];
        }
#pragma unroll
        for (int sn = 0; sn < 4; ++sn) {
            const int row = wn + sn * 16 + l15;
            b[sn].u[0] = *(const v4u*)&ws[buf][row][kb];
            b[sn].u[1] = *(const v4u*)&ws[buf][row][kb + 16];
        }

        // ---- 16 WMMAs per wave per K-step ----
#pragma unroll
        for (int sm = 0; sm < 4; ++sm)
#pragma unroll
            for (int sn = 0; sn < 4; ++sn)
                acc[sm][sn] = __builtin_amdgcn_wmma_f32_16x16x32_bf16(
                    false, a[sm].f, false, b[sn].f,
                    (short)0, acc[sm][sn], false, false);

        // ---- convert + store next tile into the other buffer ----
        if (has_next) {
            const int nb = buf ^ 1;
#pragma unroll
            for (int p = 0; p < 8; ++p) {
                v2u pk;
                pk.x = pack2(xv[p].x, xv[p].y);
                pk.y = pack2(xv[p].z, xv[p].w);
                *(v2u*)&xs[nb][xr + p * 32][xc] = pk;
            }
#pragma unroll
            for (int p = 0; p < 8; ++p) {
                const int n = wr + p * 16;
                const int code = cd[p] & 0xFF;
                *(unsigned*)&ws[nb][n][wc * 2] =
                    (unsigned)gs[n][code >> 4] | ((unsigned)gs[n][code & 15] << 16);
            }
            xnext += BK;
            cnext += BK / 2;
        }

        __syncthreads();
        buf ^= 1;
    }

    // ---- store: C/D layout -> VGPR i holds M = i + 8*half, N = l15 ----
#pragma unroll
    for (int sm = 0; sm < 4; ++sm) {
        const int mbase = block_m + wm + sm * 16 + half * 8;
#pragma unroll
        for (int sn = 0; sn < 4; ++sn) {
            const int n = block_n + wn + sn * 16 + l15;
#pragma unroll
            for (int i = 0; i < 8; ++i)
                out[(size_t)(mbase + i) * NDIM + n] = acc[sm][sn][i];
        }
    }
}

extern "C" void kernel_launch(void* const* d_in, const int* in_sizes, int n_in,
                              void* d_out, int out_size, void* d_ws, size_t ws_size,
                              hipStream_t stream) {
    (void)in_sizes; (void)n_in; (void)out_size; (void)d_ws; (void)ws_size;
    const float* x          = (const float*)d_in[0];
    const float* quant_grid = (const float*)d_in[1];
    const int*   codes      = (const int*)d_in[2];
    float*       out        = (float*)d_out;

    dim3 grid(NDIM / BN, MDIM / BM);   // (32, 32)
    sub4bit_linear_wmma<<<grid, NTHREADS, 0, stream>>>(x, quant_grid, codes, out);
}
